// HSL_Block_3599182594868
// MI455X (gfx1250) — compile-verified, matrix operand order checked by reading, not run on hardware
//
#include <hip/hip_runtime.h>

// ---------------------------------------------------------------------------
// CDNA5 / gfx1250 implementation of the HSL block.
// All GEMM-shaped work lowered to v_wmma_f32_16x16x32_f16 (wave32, one wave
// per block, EXEC all ones). 64x32 register-blocked tiles: 8 accumulators,
// A fragments reused 2x, B fragments 4x, 8 back-to-back WMMAs per K-chunk.
// ---------------------------------------------------------------------------

typedef __attribute__((ext_vector_type(16))) _Float16 v16h;
typedef __attribute__((ext_vector_type(8)))  float    v8f;

union U16h { v16h v; _Float16 e[16]; };
union U8f  { v8f  v; float    e[8]; };

static constexpr int kB  = 2;
static constexpr int kE  = 24;
static constexpr int kS  = 13824;   // 24^3 spatial
static constexpr int kL  = 13824;   // sequence length
static constexpr int kC  = 32;      // half dim
static constexpr int kDI = 64;      // mamba inner dim
static constexpr int kDS = 16;      // state dim

static __device__ __forceinline__ float geluf(float x) {
  return 0.5f * x * (1.0f + erff(x * 0.70710678118654752f));
}
static __device__ __forceinline__ float siluf(float x) {
  return x / (1.0f + expf(-x));
}
static __device__ __forceinline__ float softplusf(float x) {
  return (x > 20.0f) ? x : log1pf(expf(x));
}

// ---------------------------------------------------------------------------
// Generic strided GEMM:  Out[b,m,n] = act( sum_k A[b,m,k] * W[b,n,k] + bias[n]
//                                          (+ resid[b,m,n]) )
// M multiple of 64, N multiple of 32, K multiple of 32.
// One wave computes a 64x32 output tile (4x2 grid of 16x16 WMMA subtiles).
// ---------------------------------------------------------------------------
__launch_bounds__(32)
__global__ void wmma_gemm(const float* __restrict__ A, const float* __restrict__ W,
                          const float* __restrict__ bias, const float* __restrict__ resid,
                          float* __restrict__ Out, int K,
                          long long a_bs, long long a_ts, long long a_ks,
                          long long w_bs, long long w_ns, long long w_ks,
                          long long o_bs, long long o_ts, long long o_ns,
                          long long r_bs, long long r_ts, long long r_ns,
                          int act)
{
  __shared__ _Float16 As[64 * 32];
  __shared__ _Float16 Bs[32 * 32];
  const int lane = threadIdx.x;
  const long long m0 = (long long)blockIdx.x * 64;
  const long long n0 = (long long)blockIdx.y * 32;
  const int b = blockIdx.z;
  const float* Ab = A + (long long)b * a_bs;
  const float* Wb = W + (long long)b * w_bs;

  v8f acc[4][2];
  const v8f vzero = {0.f, 0.f, 0.f, 0.f, 0.f, 0.f, 0.f, 0.f};
#pragma unroll
  for (int mi = 0; mi < 4; ++mi)
#pragma unroll
    for (int ni = 0; ni < 2; ++ni) acc[mi][ni] = vzero;

  for (int k0 = 0; k0 < K; k0 += 32) {
    if (k0 + 32 < K)
      __builtin_prefetch(&Wb[(n0 + lane) * w_ns + (long long)(k0 + 32) * w_ks], 0, 1);
    // Stage A (64 rows x 32 k): two rows per lane.
#pragma unroll
    for (int half = 0; half < 2; ++half) {
      const int r = lane * 2 + half;
      const float* ap = Ab + (m0 + r) * a_ts;
#pragma unroll 8
      for (int i = 0; i < 32; ++i)
        As[r * 32 + i] = (_Float16)ap[(long long)(k0 + i) * a_ks];
    }
    // Stage B (32 n x 32 k): one n-row per lane.
    {
      const float* wp = Wb + (n0 + lane) * w_ns;
#pragma unroll 8
      for (int i = 0; i < 32; ++i)
        Bs[lane * 32 + i] = (_Float16)wp[(long long)(k0 + i) * w_ks];
    }
    __syncthreads();
    // gfx1250 16-bit fragment layout: lane<16 -> K{0..7,16..23}, lane>=16 -> K{8..15,24..31}
    const int row = lane & 15;
    const int kb  = (lane < 16) ? 0 : 8;
    U16h au[4], bu[2];
#pragma unroll
    for (int mi = 0; mi < 4; ++mi)
#pragma unroll
      for (int i = 0; i < 8; ++i) {
        au[mi].e[i]     = As[(mi * 16 + row) * 32 + kb + i];
        au[mi].e[i + 8] = As[(mi * 16 + row) * 32 + kb + 16 + i];
      }
#pragma unroll
    for (int ni = 0; ni < 2; ++ni)
#pragma unroll
      for (int i = 0; i < 8; ++i) {
        bu[ni].e[i]     = Bs[(ni * 16 + row) * 32 + kb + i];
        bu[ni].e[i + 8] = Bs[(ni * 16 + row) * 32 + kb + 16 + i];
      }
#pragma unroll
    for (int mi = 0; mi < 4; ++mi)
#pragma unroll
      for (int ni = 0; ni < 2; ++ni)
        acc[mi][ni] = __builtin_amdgcn_wmma_f32_16x16x32_f16(
            false, au[mi].v, false, bu[ni].v, (short)0, acc[mi][ni], false, false);
    __syncthreads();
  }

  // C/D layout: VGPR r: lanes 0..15 -> M=r, lanes 16..31 -> M=r+8; N = lane&15
  const int col  = lane & 15;
  const int rofs = (lane < 16) ? 0 : 8;
#pragma unroll
  for (int mi = 0; mi < 4; ++mi)
#pragma unroll
    for (int ni = 0; ni < 2; ++ni) {
      U8f cu; cu.v = acc[mi][ni];
#pragma unroll
      for (int r = 0; r < 8; ++r) {
        const long long mm = m0 + mi * 16 + rofs + r;
        const long long nn = n0 + ni * 16 + col;
        float val = cu.e[r];
        if (bias)  val += bias[nn];
        if (resid) val += resid[(long long)b * r_bs + mm * r_ts + nn * r_ns];
        if (act == 1)      val = geluf(val);
        else if (act == 2) val = siluf(val);
        Out[(long long)b * o_bs + mm * o_ts + nn * o_ns] = val;
      }
    }
}

// ---------------------------------------------------------------------------
// Implicit-GEMM 3x3x3 conv (pad 1) with fused bias + BN + GELU.
// X: (B, Cin, S) channel-major.  Wt: (64, Cin, 27).  Out: (B, 64, S).
// 64 voxels x 32 outputs per wave, K = 27 taps x Cin.
// ---------------------------------------------------------------------------
__launch_bounds__(32)
__global__ void conv3_bn_gelu_wmma(const float* __restrict__ X, const float* __restrict__ Wt,
                                   const float* __restrict__ cb,
                                   const float* __restrict__ bg, const float* __restrict__ bbt,
                                   const float* __restrict__ bm, const float* __restrict__ bv,
                                   float* __restrict__ Out, int Cin)
{
  __shared__ _Float16 As[64 * 32];
  __shared__ _Float16 Bs[32 * 32];
  const int lane = threadIdx.x;
  const int m0 = blockIdx.x * 64;
  const int n0 = blockIdx.y * 32;
  const int b  = blockIdx.z;
  const float* Xb = X + (long long)b * Cin * kS;

  // two voxel rows per lane
  const int s0 = m0 + lane * 2;
  const int s1 = s0 + 1;
  const int wv0 = s0 % kE, hv0 = (s0 / kE) % kE, dv0 = s0 / (kE * kE);
  const int wv1 = s1 % kE, hv1 = (s1 / kE) % kE, dv1 = s1 / (kE * kE);

  v8f acc[4][2];
  const v8f vzero = {0.f, 0.f, 0.f, 0.f, 0.f, 0.f, 0.f, 0.f};
#pragma unroll
  for (int mi = 0; mi < 4; ++mi)
#pragma unroll
    for (int ni = 0; ni < 2; ++ni) acc[mi][ni] = vzero;

  for (int tap = 0; tap < 27; ++tap) {
    const int kd = tap / 9, kh = (tap / 3) % 3, kw = tap % 3;
    const int dd0 = dv0 + kd - 1, hh0 = hv0 + kh - 1, ww0 = wv0 + kw - 1;
    const int dd1 = dv1 + kd - 1, hh1 = hv1 + kh - 1, ww1 = wv1 + kw - 1;
    const bool ok0 = (unsigned)dd0 < (unsigned)kE && (unsigned)hh0 < (unsigned)kE &&
                     (unsigned)ww0 < (unsigned)kE;
    const bool ok1 = (unsigned)dd1 < (unsigned)kE && (unsigned)hh1 < (unsigned)kE &&
                     (unsigned)ww1 < (unsigned)kE;
    const int svox0 = (dd0 * kE + hh0) * kE + ww0;
    const int svox1 = (dd1 * kE + hh1) * kE + ww1;

    for (int kc = 0; kc < Cin; kc += 32) {
#pragma unroll 8
      for (int i = 0; i < 32; ++i) {
        const long long cs = (long long)(kc + i) * kS;
        As[(lane * 2 + 0) * 32 + i] = ok0 ? (_Float16)Xb[cs + svox0] : (_Float16)0.0f;
        As[(lane * 2 + 1) * 32 + i] = ok1 ? (_Float16)Xb[cs + svox1] : (_Float16)0.0f;
      }
      {
        const float* wp = Wt + (long long)(n0 + lane) * Cin * 27 + tap;
#pragma unroll 8
        for (int i = 0; i < 32; ++i)
          Bs[lane * 32 + i] = (_Float16)wp[(long long)(kc + i) * 27];
      }
      __syncthreads();
      const int row = lane & 15;
      const int kb  = (lane < 16) ? 0 : 8;
      U16h au[4], bu[2];
#pragma unroll
      for (int mi = 0; mi < 4; ++mi)
#pragma unroll
        for (int i = 0; i < 8; ++i) {
          au[mi].e[i]     = As[(mi * 16 + row) * 32 + kb + i];
          au[mi].e[i + 8] = As[(mi * 16 + row) * 32 + kb + 16 + i];
        }
#pragma unroll
      for (int ni = 0; ni < 2; ++ni)
#pragma unroll
        for (int i = 0; i < 8; ++i) {
          bu[ni].e[i]     = Bs[(ni * 16 + row) * 32 + kb + i];
          bu[ni].e[i + 8] = Bs[(ni * 16 + row) * 32 + kb + 16 + i];
        }
#pragma unroll
      for (int mi = 0; mi < 4; ++mi)
#pragma unroll
        for (int ni = 0; ni < 2; ++ni)
          acc[mi][ni] = __builtin_amdgcn_wmma_f32_16x16x32_f16(
              false, au[mi].v, false, bu[ni].v, (short)0, acc[mi][ni], false, false);
      __syncthreads();
    }
  }

  const int col  = lane & 15;
  const int rofs = (lane < 16) ? 0 : 8;
#pragma unroll
  for (int mi = 0; mi < 4; ++mi)
#pragma unroll
    for (int ni = 0; ni < 2; ++ni) {
      U8f cu; cu.v = acc[mi][ni];
#pragma unroll
      for (int r = 0; r < 8; ++r) {
        const long long mm = m0 + mi * 16 + rofs + r;
        const int nn = n0 + ni * 16 + col;
        float y = cu.e[r] + cb[nn];
        y = (y - bm[nn]) * rsqrtf(bv[nn] + 1e-5f) * bg[nn] + bbt[nn];
        Out[(long long)b * 64 * kS + (long long)nn * kS + mm] = geluf(y);
      }
    }
}

// ---------------------------------------------------------------------------
// Depthwise dilated 3D conv (attconv). X: channel-major slice, b-stride xbs.
// ---------------------------------------------------------------------------
__global__ void dwconv3d(const float* __restrict__ X, long long xbs,
                         const float* __restrict__ Wd, const float* __restrict__ bd,
                         float* __restrict__ Y, long long ybs,
                         int Kk, int pad, int dil)
{
  long long idx = (long long)blockIdx.x * blockDim.x + threadIdx.x;
  const long long total = (long long)kB * kC * kS;
  if (idx >= total) return;
  const int s = (int)(idx % kS);
  const int c = (int)((idx / kS) % kC);
  const int b = (int)(idx / ((long long)kS * kC));
  const int wv = s % kE, hv = (s / kE) % kE, dv = s / (kE * kE);
  const float* xp = X + (long long)b * xbs + (long long)c * kS;
  const float* wp = Wd + (long long)c * Kk * Kk * Kk;
  float acc = bd[c];
  for (int kd = 0; kd < Kk; ++kd) {
    const int dd = dv + kd * dil - pad; if ((unsigned)dd >= (unsigned)kE) continue;
    for (int kh = 0; kh < Kk; ++kh) {
      const int hh = hv + kh * dil - pad; if ((unsigned)hh >= (unsigned)kE) continue;
      for (int kw = 0; kw < Kk; ++kw) {
        const int ww = wv + kw * dil - pad; if ((unsigned)ww >= (unsigned)kE) continue;
        acc += wp[(kd * Kk + kh) * Kk + kw] * xp[(dd * kE + hh) * kE + ww];
      }
    }
  }
  Y[(long long)b * ybs + (long long)c * kS + s] = acc;
}

// LayerNorm over C=32 channels: xa (B,32,S) channel-major -> xn (B,L,32) token-major
__global__ void ln_kernel(const float* __restrict__ xa, const float* __restrict__ g,
                          const float* __restrict__ bt, float* __restrict__ xn)
{
  long long idx = (long long)blockIdx.x * blockDim.x + threadIdx.x;
  if (idx >= (long long)kB * kS) return;
  const int s = (int)(idx % kS);
  const int b = (int)(idx / kS);
  const float* xp = xa + (long long)b * kC * kS + s;
  float m = 0.f;
  for (int c = 0; c < kC; ++c) m += xp[(long long)c * kS];
  m /= (float)kC;
  float v = 0.f;
  for (int c = 0; c < kC; ++c) { float d = xp[(long long)c * kS] - m; v += d * d; }
  v /= (float)kC;
  const float inv = rsqrtf(v + 1e-5f);
  float* o = xn + ((long long)b * kL + s) * kC;
  for (int c = 0; c < kC; ++c) o[c] = (xp[(long long)c * kS] - m) * inv * g[c] + bt[c];
}

// Causal depthwise conv1d (k=4, left pad 3) + SiLU. src token stride ts.
__global__ void cconv1d_silu(const float* __restrict__ src, long long ts, long long bs,
                             const float* __restrict__ w, const float* __restrict__ cb,
                             float* __restrict__ dst)
{
  long long idx = (long long)blockIdx.x * blockDim.x + threadIdx.x;
  if (idx >= (long long)kB * kL * kDI) return;
  const int d = (int)(idx % kDI);
  const int t = (int)((idx / kDI) % kL);
  const int b = (int)(idx / ((long long)kDI * kL));
  const float* sp = src + (long long)b * bs + d;
  float a = cb[d];
  for (int j = 0; j < 4; ++j) {
    const int tt = t - 3 + j;
    if (tt >= 0) a += w[d * 4 + j] * sp[(long long)tt * ts];
  }
  dst[((long long)b * kL + t) * kDI + d] = siluf(a);
}

// reverse xi slice of xz (token stride 128) into contiguous (B,L,64)
__global__ void revcopy(const float* __restrict__ xz, float* __restrict__ dst)
{
  long long idx = (long long)blockIdx.x * blockDim.x + threadIdx.x;
  if (idx >= (long long)kB * kL * kDI) return;
  const int d = (int)(idx % kDI);
  const int t = (int)((idx / kDI) % kL);
  const int b = (int)(idx / ((long long)kDI * kL));
  dst[((long long)b * kL + t) * kDI + d] =
      xz[((long long)b * kL + (kL - 1 - t)) * 128 + d];
}

// dbl[b,t,o] = sum_d xct[b,t,d] * xw[o,d]   (o < 34)
__global__ void xproj_kernel(const float* __restrict__ xct, const float* __restrict__ xw,
                             float* __restrict__ dbl)
{
  long long idx = (long long)blockIdx.x * blockDim.x + threadIdx.x;
  if (idx >= (long long)kB * kL * 34) return;
  const int o = (int)(idx % 34);
  const long long tb = idx / 34;
  const float* xp = xct + tb * kDI;
  float a = 0.f;
  for (int d = 0; d < kDI; ++d) a += xp[d] * xw[o * kDI + d];
  dbl[tb * 34 + o] = a;
}

// dt[b,t,d] = softplus(dbl0*dw[d,0] + dbl1*dw[d,1] + db[d])
__global__ void dt_kernel(const float* __restrict__ dbl, const float* __restrict__ dw,
                          const float* __restrict__ db, float* __restrict__ dt)
{
  long long idx = (long long)blockIdx.x * blockDim.x + threadIdx.x;
  if (idx >= (long long)kB * kL * kDI) return;
  const int d = (int)(idx % kDI);
  const long long tb = idx / kDI;
  const float* dp = dbl + tb * 34;
  dt[tb * kDI + d] = softplusf(dp[0] * dw[d * 2] + dp[1] * dw[d * 2 + 1] + db[d]);
}

// Selective scan: one thread per (b,d); 16 states in registers.
__global__ void scan_kernel(const float* __restrict__ u, const float* __restrict__ dt,
                            const float* __restrict__ dbl, const float* __restrict__ Alog,
                            const float* __restrict__ Dp, float* __restrict__ y)
{
  const int idx = blockIdx.x * blockDim.x + threadIdx.x;
  if (idx >= kB * kDI) return;
  const int d = idx % kDI;
  const int b = idx / kDI;
  float A[kDS], h[kDS];
#pragma unroll
  for (int s2 = 0; s2 < kDS; ++s2) { A[s2] = -expf(Alog[d * kDS + s2]); h[s2] = 0.f; }
  const float Dd = Dp[d];
  for (int t = 0; t < kL; ++t) {
    const long long tb = (long long)b * kL + t;
    const float dtv = dt[tb * kDI + d];
    const float uv  = u[tb * kDI + d];
    const float du  = dtv * uv;
    const float* dp = dbl + tb * 34;
    float acc = 0.f;
#pragma unroll
    for (int s2 = 0; s2 < kDS; ++s2) {
      h[s2] = expf(dtv * A[s2]) * h[s2] + du * dp[2 + s2];
      acc += h[s2] * dp[18 + s2];
    }
    y[tb * kDI + d] = acc + uv * Dd;
  }
}

// v = (yf + reverse(yb)) * silu(z);  also emit k = silu(z)
__global__ void combine_kernel(const float* __restrict__ yf, const float* __restrict__ yb,
                               const float* __restrict__ xz, float* __restrict__ vout,
                               float* __restrict__ kout)
{
  long long idx = (long long)blockIdx.x * blockDim.x + threadIdx.x;
  if (idx >= (long long)kB * kL * kDI) return;
  const int d = (int)(idx % kDI);
  const int t = (int)((idx / kDI) % kL);
  const int b = (int)(idx / ((long long)kDI * kL));
  const float kk = siluf(xz[((long long)b * kL + t) * 128 + 64 + d]);
  const float v = (yf[((long long)b * kL + t) * kDI + d] +
                   yb[((long long)b * kL + (kL - 1 - t)) * kDI + d]) * kk;
  vout[((long long)b * kL + t) * kDI + d] = v;
  kout[((long long)b * kL + t) * kDI + d] = kk;
}

// channel mean/max over att = [att1(32ch) ; att2(32ch)] -> pooled (B,2,S)
__global__ void pool_kernel(const float* __restrict__ att1, long long a1bs,
                            const float* __restrict__ att2, float* __restrict__ pooled)
{
  long long idx = (long long)blockIdx.x * blockDim.x + threadIdx.x;
  if (idx >= (long long)kB * kS) return;
  const int s = (int)(idx % kS);
  const int b = (int)(idx / kS);
  const float* p1 = att1 + (long long)b * a1bs + s;
  const float* p2 = att2 + (long long)b * kC * kS + s;
  float mean = 0.f, mx = -3.4e38f;
  for (int c = 0; c < kC; ++c) { float v = p1[(long long)c * kS]; mean += v; mx = fmaxf(mx, v); }
  for (int c = 0; c < kC; ++c) { float v = p2[(long long)c * kS]; mean += v; mx = fmaxf(mx, v); }
  pooled[((long long)b * 2 + 0) * kS + s] = mean / 64.f;
  pooled[((long long)b * 2 + 1) * kS + s] = mx;
}

// SE conv 2->2, k=7, pad 3, then sigmoid
__global__ void seconv_kernel(const float* __restrict__ pooled, const float* __restrict__ sw,
                              const float* __restrict__ sb, float* __restrict__ se)
{
  long long idx = (long long)blockIdx.x * blockDim.x + threadIdx.x;
  if (idx >= (long long)kB * 2 * kS) return;
  const int s  = (int)(idx % kS);
  const int oc = (int)((idx / kS) % 2);
  const int b  = (int)(idx / ((long long)kS * 2));
  const int wv = s % kE, hv = (s / kE) % kE, dv = s / (kE * kE);
  float acc = sb[oc];
  for (int ic = 0; ic < 2; ++ic) {
    const float* xp = pooled + ((long long)b * 2 + ic) * kS;
    const float* wp = sw + ((long long)oc * 2 + ic) * 343;
    for (int kd = 0; kd < 7; ++kd) {
      const int dd = dv + kd - 3; if ((unsigned)dd >= (unsigned)kE) continue;
      for (int kh = 0; kh < 7; ++kh) {
        const int hh = hv + kh - 3; if ((unsigned)hh >= (unsigned)kE) continue;
        for (int kw = 0; kw < 7; ++kw) {
          const int ww = wv + kw - 3; if ((unsigned)ww >= (unsigned)kE) continue;
          acc += wp[(kd * 7 + kh) * 7 + kw] * xp[(dd * kE + hh) * kE + ww];
        }
      }
    }
  }
  se[((long long)b * 2 + oc) * kS + s] = 1.0f / (1.0f + expf(-acc));
}

// hsl output = att1*se0 + att2*se1, written into cat slice (b-stride 128*S)
__global__ void hslout_kernel(const float* __restrict__ att1, long long a1bs,
                              const float* __restrict__ att2, const float* __restrict__ se,
                              float* __restrict__ catdst, long long cbs)
{
  long long idx = (long long)blockIdx.x * blockDim.x + threadIdx.x;
  if (idx >= (long long)kB * kC * kS) return;
  const int s = (int)(idx % kS);
  const int c = (int)((idx / kS) % kC);
  const int b = (int)(idx / ((long long)kS * kC));
  const float a1 = att1[(long long)b * a1bs + (long long)c * kS + s];
  const float a2 = att2[(long long)b * kC * kS + (long long)c * kS + s];
  const float s0 = se[((long long)b * 2 + 0) * kS + s];
  const float s1 = se[((long long)b * 2 + 1) * kS + s];
  catdst[(long long)b * cbs + (long long)c * kS + s] = a1 * s0 + a2 * s1;
}

// row softmax over 64 entries (B*64 rows)
__global__ void softmax64(float* __restrict__ logits)
{
  const int idx = blockIdx.x * blockDim.x + threadIdx.x;
  if (idx >= kB * 64) return;
  float* row = logits + (long long)idx * 64;
  float mx = -3.4e38f;
  for (int n = 0; n < 64; ++n) mx = fmaxf(mx, row[n]);
  float sum = 0.f;
  for (int n = 0; n < 64; ++n) { row[n] = expf(row[n] - mx); sum += row[n]; }
  const float inv = 1.0f / sum;
  for (int n = 0; n < 64; ++n) row[n] *= inv;
}

// ---------------------------------------------------------------------------
extern "C" void kernel_launch(void* const* d_in, const int* in_sizes, int n_in,
                              void* d_out, int out_size, void* d_ws, size_t ws_size,
                              hipStream_t stream)
{
  (void)in_sizes; (void)out_size; (void)ws_size;
  const int B = kB, S = kS, L = kL, C = kC, DI = kDI;
  const float* in[65];
  for (int i = 0; i < 65 && i < n_in; ++i) in[i] = (const float*)d_in[i];

  float* ws = (float*)d_ws;
  size_t off = 0;
  auto alloc = [&](size_t n) { float* p = ws + off; off += n; return p; };
  float* f_h    = alloc((size_t)B * 64 * S);   // proj1 output (B,64,S) channel-major
  float* xa     = alloc((size_t)B * C  * S);
  float* xn     = alloc((size_t)B * L  * C);
  float* xz     = alloc((size_t)B * L  * 128);
  float* xctf   = alloc((size_t)B * L  * DI);
  float* dblf   = alloc((size_t)B * L  * 34);
  float* dtf    = alloc((size_t)B * L  * DI);
  float* yf     = alloc((size_t)B * L  * DI);
  float* xirev  = alloc((size_t)B * L  * DI);
  float* xctb   = alloc((size_t)B * L  * DI);
  float* dblb   = alloc((size_t)B * L  * 34);
  float* dtb    = alloc((size_t)B * L  * DI);
  float* yb     = alloc((size_t)B * L  * DI);
  float* vbuf   = alloc((size_t)B * L  * DI);
  float* kbuf   = alloc((size_t)B * L  * DI);
  float* v_g    = alloc((size_t)B * L  * DI);
  float* k_g    = alloc((size_t)B * L  * DI);
  float* att2   = alloc((size_t)B * C  * S);
  float* pooled = alloc((size_t)B * 2  * S);
  float* sebuf  = alloc((size_t)B * 2  * S);
  float* catb   = alloc((size_t)B * 128 * S);
  float* c1o    = alloc((size_t)B * 64 * S);
  float* c2o    = alloc((size_t)B * 64 * S);
  float* p2o    = alloc((size_t)B * 64 * S);
  float* mlp1   = alloc((size_t)B * 128 * S);
  float* logits = alloc((size_t)B * 64 * 64);

  auto gemm = [&](const float* A, const float* W, const float* bias, const float* resid,
                  float* O, int M, int N, int K, int Bz,
                  long long abs_, long long ats, long long aks,
                  long long wbs, long long wns, long long wks,
                  long long obs, long long ots, long long ons,
                  long long rbs, long long rts, long long rns, int act) {
    dim3 g((unsigned)(M / 64), (unsigned)(N / 32), (unsigned)Bz);
    wmma_gemm<<<g, 32, 0, stream>>>(A, W, bias, resid, O, K, abs_, ats, aks,
                                    wbs, wns, wks, obs, ots, ons, rbs, rts, rns, act);
  };
  auto blocks = [](long long n) { return dim3((unsigned)((n + 255) / 256)); };

  // ---- proj1 + GELU: (B,64,S) tokens = spatial, K over channels (stride S)
  gemm(in[0], in[1], in[2], nullptr, f_h, S, 64, 64, B,
       (long long)64 * S, 1, S,  0, 64, 1,  (long long)64 * S, 1, S,  0, 0, 0, 1);

  // ---- two HSL instances (0 = global on channels 0..31, 1 = local on 32..63)
  auto run_hsl = [&](int inst) {
    const float* const* P = in + (inst == 0 ? 9 : 31);
    const int kk  = inst == 0 ? 7 : 5;
    const int pad = inst == 0 ? 9 : 2;
    const int dil = inst == 0 ? 3 : 1;
    const float* att1 = f_h + (size_t)inst * C * S;      // b-stride 64*S

    dwconv3d<<<blocks((long long)B * C * S), 256, 0, stream>>>(
        att1, (long long)64 * S, P[0], P[1], xa, (long long)C * S, kk, pad, dil);
    ln_kernel<<<blocks((long long)B * S), 256, 0, stream>>>(xa, P[4], P[5], xn);

    // in_proj: (B,L,32) @ (128,32)^T -> xz (B,L,128)
    gemm(xn, P[6], nullptr, nullptr, xz, L, 128, 32, B,
         (long long)L * 32, 32, 1,  0, 32, 1,  (long long)L * 128, 128, 1,  0, 0, 0, 0);

    // forward branch
    cconv1d_silu<<<blocks((long long)B * L * DI), 256, 0, stream>>>(
        xz, 128LL, (long long)L * 128, P[7], P[8], xctf);
    xproj_kernel<<<blocks((long long)B * L * 34), 256, 0, stream>>>(xctf, P[9], dblf);
    dt_kernel<<<blocks((long long)B * L * DI), 256, 0, stream>>>(dblf, P[10], P[11], dtf);
    scan_kernel<<<1, 128, 0, stream>>>(xctf, dtf, dblf, P[12], P[13], yf);

    // backward branch on reversed sequence
    revcopy<<<blocks((long long)B * L * DI), 256, 0, stream>>>(xz, xirev);
    cconv1d_silu<<<blocks((long long)B * L * DI), 256, 0, stream>>>(
        xirev, 64LL, (long long)L * 64, P[14], P[15], xctb);
    xproj_kernel<<<blocks((long long)B * L * 34), 256, 0, stream>>>(xctb, P[16], dblb);
    dt_kernel<<<blocks((long long)B * L * DI), 256, 0, stream>>>(dblb, P[17], P[18], dtb);
    scan_kernel<<<1, 128, 0, stream>>>(xctb, dtb, dblb, P[19], P[20], yb);

    float* vdst = inst == 0 ? v_g : vbuf;
    float* kdst = inst == 0 ? k_g : kbuf;
    combine_kernel<<<blocks((long long)B * L * DI), 256, 0, stream>>>(yf, yb, xz, vdst, kdst);

    // out_proj: v (B,L,64) @ (32,64)^T -> att2 (B,32,S) channel-major
    gemm(vdst, P[21], nullptr, nullptr, att2, L, 32, 64, B,
         (long long)L * 64, 64, 1,  0, 64, 1,  (long long)C * S, 1, (long long)S,  0, 0, 0, 0);

    pool_kernel<<<blocks((long long)B * S), 256, 0, stream>>>(att1, (long long)64 * S, att2, pooled);
    seconv_kernel<<<blocks((long long)B * 2 * S), 256, 0, stream>>>(pooled, P[2], P[3], sebuf);
    hslout_kernel<<<blocks((long long)B * C * S), 256, 0, stream>>>(
        att1, (long long)64 * S, att2, sebuf, catb + (size_t)inst * C * S, (long long)128 * S);
  };
  run_hsl(0);
  run_hsl(1);   // leaves q_l in xctf

  // ---- channel attention: logits[b] = q_l^T @ k_g  (64x64, K = L)
  gemm(xctf, k_g, nullptr, nullptr, logits, 64, 64, L, B,
       (long long)L * 64, 1, 64,  (long long)L * 64, 1, 64,  4096, 64, 1,  0, 0, 0, 0);
  softmax64<<<1, 128, 0, stream>>>(logits);
  // out_a = v_g @ attn^T, stored token-major => matches reference's reshape to (B,64,D,H,W)
  gemm(v_g, logits, nullptr, nullptr, catb + (size_t)64 * S, L, 64, 64, B,
       (long long)L * 64, 64, 1,  4096, 64, 1,  (long long)128 * S, 64, 1,  0, 0, 0, 0);

  // ---- conv block (implicit GEMM, fused BN+GELU)
  const float* const* D = in + 53;
  conv3_bn_gelu_wmma<<<dim3(S / 64, 2, B), 32, 0, stream>>>(
      catb, D[0], D[1], D[2], D[3], D[4], D[5], c1o, 128);
  conv3_bn_gelu_wmma<<<dim3(S / 64, 2, B), 32, 0, stream>>>(
      c1o, D[6], D[7], D[8], D[9], D[10], D[11], c2o, 64);

  // ---- proj2 + residual (sc = input x)
  gemm(c2o, in[3], in[4], in[0], p2o, S, 64, 64, B,
       (long long)64 * S, 1, S,  0, 64, 1,  (long long)64 * S, 1, S,
       (long long)64 * S, 1, (long long)S, 0);

  // ---- MLP: w1 (64->128, GELU), w2 (128->64) -> d_out (NCDHW)
  gemm(p2o, in[5], in[6], nullptr, mlp1, S, 128, 64, B,
       (long long)64 * S, 1, S,  0, 64, 1,  (long long)128 * S, 1, S,  0, 0, 0, 1);
  gemm(mlp1, in[7], in[8], nullptr, (float*)d_out, S, 64, 128, B,
       (long long)128 * S, 1, S,  0, 128, 1,  (long long)64 * S, 1, S,  0, 0, 0, 0);
}
